// TransformerCNN_25769803969
// MI455X (gfx1250) — matrix operand order repaired and checked by reading, not compiled, not run on hardware
//
#include <hip/hip_runtime.h>
#include <cstdint>
#include <cstddef>

// ---------------------------------------------------------------------------
// TransformerCNN for MI455X (gfx1250): bf16 implicit-GEMM convs on WMMA.
//   Heavy ops: 5x5 cin->cin convs (~3.4 TFLOP) -> v_wmma_f32_16x16x32_bf16.
//   Activations: spatially zero-padded NHWC bf16 (20x28 frame per 16x24 image)
//     -> every 5x5 tap load is in-bounds; all tap/kc offsets are immediates.
//   Weights: tap-major bf16 [co][t][ci]; per-tap 32xCp slice staged in LDS
//     via global_load_async_to_lds_b128 (ASYNCcnt), ping-pong buffered.
//   Register blocking: each wave computes 32cout x 32pix (2x2 WMMA tiles):
//     2 A-frags + 2 B-frags -> 4 WMMAs, i.e. 0.5KB global + 0.5KB LDS / WMMA.
// ---------------------------------------------------------------------------

typedef __bf16 bf16;
typedef __attribute__((ext_vector_type(16))) __bf16 v16bf;
typedef __attribute__((ext_vector_type(8)))  float  v8f;
typedef __attribute__((ext_vector_type(4)))  unsigned int u32x4;
typedef __attribute__((ext_vector_type(4)))  int          i32x4;

#define NB   64
#define CH   256
#define HT   16
#define WD   24
#define HW   384          // HT*WD
#define PH   20           // HT + 4 (padded frame)
#define PW   28           // WD + 4
#define PHW  560          // PH*PW
#define KK   10
#define BK   640          // NB*KK
#define CP_A 288          // padded channels for a1/a2 (258/259 -> 288)

#if defined(__has_builtin)
#  if __has_builtin(__builtin_amdgcn_global_load_async_to_lds_b128) && \
      __has_builtin(__builtin_amdgcn_s_wait_asynccnt)
#    define HAVE_ASYNC_LDS 1
#  endif
#endif

#ifdef HAVE_ASYNC_LDS
// Builtin signature (probe-confirmed): (v4i AS1*, v4i AS3*, Ii, Ii)
#define AS_GLOBAL_V4(p) ((__attribute__((address_space(1))) i32x4*)(p))
#define AS_SHARED_V4(p) ((__attribute__((address_space(3))) i32x4*)(p))
#endif

__device__ __forceinline__ float logsig(float x) {
  return fminf(x, 0.f) - log1pf(expf(-fabsf(x)));   // stable log(sigmoid(x))
}

// ---------------------------------------------------------------------------
// 5x5 conv, cin(Cp)->cout(Cp), padded NHWC bf16 in/out, fused bias+ReLU.
// Wave macro-tile: 32 cout x 32 pixels = 2x2 WMMA tiles, 4 f32 accumulators.
// Block: 8 waves, one 32-cout slice, 256 pixels.
// grid.x = nPix/256, grid.y = Cp/32.
// ---------------------------------------------------------------------------
template <int Cp>
__global__ void __launch_bounds__(256)
conv5x5_wmma(const bf16* __restrict__ act, const bf16* __restrict__ wgt,
             const float* __restrict__ bias, bf16* __restrict__ out,
             int coutReal) {
  __shared__ bf16 wtile[2][32 * Cp];

  const int tid   = threadIdx.x;
  const int lane  = tid & 31;
  const int wave  = tid >> 5;
  const int laneM = lane & 15;
  const int half  = lane >> 4;
  const int n0    = (blockIdx.x * 8 + wave) * 32;
  const int co0   = blockIdx.y * 32;

  // Two 16-pixel B tiles per wave.
  int pixoff[2];
  const bf16* pixbase[2];
#pragma unroll
  for (int i = 0; i < 2; ++i) {
    const int n = n0 + 16 * i + laneM;
    const int b = n / HW, r = n % HW, h = r / WD, w = r % WD;
    pixoff[i]  = b * PHW + (h + 2) * PW + (w + 2);
    // B-fragment: lane holds col N=laneM, K chunks [8h,8h+8) u [8h+16,8h+24).
    pixbase[i] = act + (size_t)pixoff[i] * Cp + 8 * half;
  }

  const bf16* wbase = wgt + (size_t)co0 * (25 * Cp);

  v8f acc[2][2];
#pragma unroll
  for (int i = 0; i < 2; ++i)
#pragma unroll
    for (int jc = 0; jc < 2; ++jc)
      acc[i][jc] = (v8f){0.f, 0.f, 0.f, 0.f, 0.f, 0.f, 0.f, 0.f};

  for (int t = 0; t < 25; ++t) {
    const int buf = t & 1;
    // ------ cooperative stage of the 32 x Cp tap-t weight slice -> LDS -----
    for (int j = tid; j < 32 * Cp / 8; j += 256) {
      const int row = j / (Cp / 8);
      const int c8  = (j % (Cp / 8)) * 8;
      const bf16* src = wbase + (size_t)row * (25 * Cp) + t * Cp + c8;
#ifdef HAVE_ASYNC_LDS
      __builtin_amdgcn_global_load_async_to_lds_b128(
          AS_GLOBAL_V4(src), AS_SHARED_V4(&wtile[buf][row * Cp + c8]), 0, 0);
#else
      *(u32x4*)&wtile[buf][row * Cp + c8] = *(const u32x4*)src;
#endif
    }
#ifdef HAVE_ASYNC_LDS
    __builtin_amdgcn_s_wait_asynccnt(0);
#endif
    __syncthreads();   // fill(t) done; also fences reads of wtile[buf] @ t-2

    const int dh = t / 5 - 2;
    const int dw = t % 5 - 2;
    const ptrdiff_t toff = (ptrdiff_t)(dh * PW + dw) * Cp;  // always in-bounds
    // A-fragments from LDS: lane row M=laneM, K chunks [8h,8h+8) u [8h+16,8h+24)
    const bf16* at0 = &wtile[buf][laneM * Cp + 8 * half];
    const bf16* at1 = at0 + 16 * Cp;

#pragma unroll
    for (int kc = 0; kc < Cp; kc += 32) {
      union { v16bf v; u32x4 q[2]; } A0, A1, B0, B1;
      A0.q[0] = *(const u32x4*)(at0 + kc);
      A0.q[1] = *(const u32x4*)(at0 + kc + 16);
      A1.q[0] = *(const u32x4*)(at1 + kc);
      A1.q[1] = *(const u32x4*)(at1 + kc + 16);
      B0.q[0] = *(const u32x4*)(pixbase[0] + toff + kc);
      B0.q[1] = *(const u32x4*)(pixbase[0] + toff + kc + 16);
      B1.q[0] = *(const u32x4*)(pixbase[1] + toff + kc);
      B1.q[1] = *(const u32x4*)(pixbase[1] + toff + kc + 16);
      acc[0][0] = __builtin_amdgcn_wmma_f32_16x16x32_bf16(false, A0.v, false, B0.v,
                                                          (short)0, acc[0][0], false, false);
      acc[0][1] = __builtin_amdgcn_wmma_f32_16x16x32_bf16(false, A1.v, false, B0.v,
                                                          (short)0, acc[0][1], false, false);
      acc[1][0] = __builtin_amdgcn_wmma_f32_16x16x32_bf16(false, A0.v, false, B1.v,
                                                          (short)0, acc[1][0], false, false);
      acc[1][1] = __builtin_amdgcn_wmma_f32_16x16x32_bf16(false, A1.v, false, B1.v,
                                                          (short)0, acc[1][1], false, false);
    }
  }

  // Epilogue: lane covers rows M = 8*half + r2 of each co-tile, col N = laneM.
#pragma unroll
  for (int i = 0; i < 2; ++i) {
#pragma unroll
    for (int jc = 0; jc < 2; ++jc) {
      union { u32x4 q; bf16 hv[8]; } pk;
#pragma unroll
      for (int r2 = 0; r2 < 8; ++r2) {
        const int co = co0 + 16 * jc + 8 * half + r2;
        float f = acc[i][jc][r2];
        if (co < coutReal) { f += bias[co]; f = fmaxf(f, 0.f); }
        else               { f = 0.f; }               // keep pad channels clean
        pk.hv[r2] = (bf16)f;
      }
      *(u32x4*)(out + (size_t)pixoff[i] * Cp + co0 + 16 * jc + 8 * half) = pk.q;
    }
  }
}

// ---------------------------------------------------------------------------
// LAN tail: conv3 (5x5 cin->1) + ReLU + residual 1x1 + final scalar affine.
// Padded inputs -> no bounds checks (halo reads are genuine zeros).
// ---------------------------------------------------------------------------
__global__ void __launch_bounds__(256)
lan_tail(const bf16* __restrict__ act2, const bf16* __restrict__ rep,
         const float* __restrict__ w3, const float* __restrict__ b3,
         const float* __restrict__ rw, const float* __restrict__ rbv,
         const float* __restrict__ lw, const float* __restrict__ lb,
         float* __restrict__ outMap, int nPix, int Cp, int cinReal) {
  const int p = blockIdx.x * 256 + threadIdx.x;
  if (p >= nPix) return;
  const int img = p / HW;
  const int r   = p % HW;
  const int h   = r / WD;
  const int w   = r % WD;
  const int pc  = img * PHW + (h + 2) * PW + (w + 2);

  float s3 = 0.f;
  for (int t = 0; t < 25; ++t) {
    const int dh = t / 5 - 2, dw = t % 5 - 2;
    const bf16* ab = act2 + (size_t)(pc + dh * PW + dw) * Cp;
    float s = 0.f;
    for (int ci = 0; ci < cinReal; ++ci) s += w3[ci * 25 + t] * (float)ab[ci];
    s3 += s;
  }
  s3 = fmaxf(s3 + b3[0], 0.f);

  const bf16* rp = rep + (size_t)pc * Cp;
  float rs = rbv[0];
  for (int ci = 0; ci < cinReal; ++ci) rs += rw[ci] * (float)rp[ci];

  outMap[p] = lw[0] * (s3 + rs) + lb[0];
}

// --------------------------- glue kernels ----------------------------------

__global__ void pack_x_k(const float* __restrict__ x, bf16* __restrict__ out) {
  const int i = blockIdx.x * 256 + threadIdx.x;     // NB*HW*CH threads
  const int ci  = i & (CH - 1);
  const int pix = i >> 8;
  const int b = pix / HW, r = pix % HW;
  const int h = r / WD, w = r % WD;
  out[(size_t)(b * PHW + (h + 2) * PW + (w + 2)) * CH + ci] =
      (bf16)x[((size_t)b * CH + ci) * HW + r];
}

__global__ void pack_w5_k(const float* __restrict__ w, bf16* __restrict__ out,
                          int coutR, int cinR, int Cp, int CoP) {
  const int i = blockIdx.x * 256 + threadIdx.x;
  const int total = CoP * 25 * Cp;
  if (i >= total) return;
  const int ci = i % Cp;
  const int rem = i / Cp;
  const int t  = rem % 25;
  const int co = rem / 25;
  float v = 0.f;
  if (co < coutR && ci < cinR) v = w[((size_t)co * cinR + ci) * 25 + t];
  out[i] = (bf16)v;   // [co][t][ci] tap-major, zero padded
}

__global__ void lsf_k(const float* __restrict__ m, float* __restrict__ o, int n) {
  const int i = blockIdx.x * 256 + threadIdx.x;
  if (i < n) o[i] = logsig(m[i]);
}

__global__ void localmax_k(const float* __restrict__ fg, int* __restrict__ peaks) {
  __shared__ float att[HW];
  __shared__ float mlm[HW];
  const int b = blockIdx.x;
  const int t = threadIdx.x;                 // 384 threads
  att[t] = fg[b * HW + t];
  __syncthreads();
  const int h = t / WD, w = t % WD;
  float mx = -__builtin_inff();
  for (int dh = -1; dh <= 1; ++dh)
    for (int dw = -1; dw <= 1; ++dw) {
      const int hh = h + dh, ww = w + dw;
      if ((unsigned)hh < (unsigned)HT && (unsigned)ww < (unsigned)WD)
        mx = fmaxf(mx, att[hh * WD + ww]);
    }
  mlm[t] = (att[t] == mx) ? att[t] : 0.f;
  __syncthreads();
  if (t == 0) {
    for (int k = 0; k < KK; ++k) {
      float best = -__builtin_inff(); int bi = 0;
      for (int i = 0; i < HW; ++i)
        if (mlm[i] > best) { best = mlm[i]; bi = i; }
      peaks[b * KK + k] = bi;
      mlm[bi] = -__builtin_inff();
    }
  }
}

__global__ void ind_k(const int* __restrict__ peaks, float* __restrict__ ind) {
  __shared__ float sh[HW];
  __shared__ float slse;
  const int bk = blockIdx.x;
  const int t  = threadIdx.x;
  const int idx = peaks[bk];
  const float xp = (float)(idx / WD), yp = (float)(idx % WD);
  const float h = (float)(t / WD), w = (float)(t % WD);
  const float lg = -((xp - h) * (xp - h)) * 0.5f - ((yp - w) * (yp - w)) * 0.5f;
  sh[t] = lg;
  __syncthreads();
  if (t == 0) {
    float m = -__builtin_inff();
    for (int i = 0; i < HW; ++i) m = fmaxf(m, sh[i]);
    float s = 0.f;
    for (int i = 0; i < HW; ++i) s += expf(sh[i] - m);
    slse = logf(s) + m;
  }
  __syncthreads();
  ind[(size_t)bk * HW + t] = lg - slse;
}

__global__ void rep1_k(const bf16* __restrict__ xn, const float* __restrict__ lsf,
                       const float* __restrict__ ind, bf16* __restrict__ rep) {
  const int i = blockIdx.x * 256 + threadIdx.x;   // BK*HW
  if (i >= BK * HW) return;
  const int pix = i % HW;
  const int bk  = i / HW;
  const int b   = bk / KK;
  const int h = pix / WD, w = pix % WD;
  const int po = (h + 2) * PW + (w + 2);
  const u32x4* s4 = (const u32x4*)(xn + ((size_t)b * PHW + po) * CH);
  bf16* dst = rep + ((size_t)bk * PHW + po) * CP_A;
  u32x4* d4 = (u32x4*)dst;
#pragma unroll 8
  for (int j = 0; j < CH / 8; ++j) d4[j] = s4[j];
  dst[256] = (bf16)lsf[b * HW + pix];
  dst[257] = (bf16)ind[(size_t)bk * HW + pix];
  for (int c = 258; c < CP_A; ++c) dst[c] = (bf16)0.f;
}

__global__ void scopesum_k(const float* __restrict__ att1, const int* __restrict__ objlen,
                           float* __restrict__ ssum) {
  const int i = blockIdx.x * 256 + threadIdx.x;   // NB*HW
  if (i >= NB * HW) return;
  const int b = i / HW, pix = i % HW;
  const int L = objlen[b];
  float s = 0.f;
  for (int k = 0; k < KK; ++k)
    if (k < L) s += logsig(-att1[((size_t)(b * KK + k)) * HW + pix]);
  ssum[i] = s;
}

__global__ void rep2_k(const bf16* __restrict__ xn, const float* __restrict__ lsf,
                       const float* __restrict__ att1, const float* __restrict__ ssum,
                       bf16* __restrict__ rep) {
  const int i = blockIdx.x * 256 + threadIdx.x;   // BK*HW
  if (i >= BK * HW) return;
  const int pix = i % HW;
  const int bk  = i / HW;
  const int b   = bk / KK;
  const int h = pix / WD, w = pix % WD;
  const int po = (h + 2) * PW + (w + 2);
  const u32x4* s4 = (const u32x4*)(xn + ((size_t)b * PHW + po) * CH);
  bf16* dst = rep + ((size_t)bk * PHW + po) * CP_A;
  u32x4* d4 = (u32x4*)dst;
#pragma unroll 8
  for (int j = 0; j < CH / 8; ++j) d4[j] = s4[j];
  const float a = att1[(size_t)bk * HW + pix];
  dst[256] = (bf16)lsf[b * HW + pix];
  dst[257] = (bf16)a;
  dst[258] = (bf16)(ssum[b * HW + pix] - logsig(-a));   // scope
  for (int c = 259; c < CP_A; ++c) dst[c] = (bf16)0.f;
}

// ---------------------------------------------------------------------------

extern "C" void kernel_launch(void* const* d_in, const int* in_sizes, int n_in,
                              void* d_out, int out_size, void* d_ws, size_t ws_size,
                              hipStream_t stream) {
  (void)in_sizes; (void)n_in; (void)out_size; (void)ws_size;

  const float* x      = (const float*)d_in[0];
  const int*   objlen = (const int*)d_in[1];
  const float* P[30];
  for (int i = 0; i < 30; ++i) P[i] = (const float*)d_in[2 + i];
  const float** FG = P;        // rw rb w1 b1 w2 b2 w3 b3 lw lb
  const float** A1 = P + 10;
  const float** A2 = P + 20;

  char* base = (char*)d_ws;
  size_t off = 0;
  auto alloc = [&](size_t bytes) -> char* {
    char* q = base + off;
    off += (bytes + 255) & ~size_t(255);
    return q;
  };

  const size_t SZ_FG  = (size_t)NB * PHW * CH * 2;     // padded NHWC, 256 ch
  const size_t SZ_BIG = (size_t)BK * PHW * CP_A * 2;   // padded NHWC, 288 ch

  bf16*  XNP   = (bf16*)alloc(SZ_FG);
  bf16*  FGA1  = (bf16*)alloc(SZ_FG);
  bf16*  FGA2  = (bf16*)alloc(SZ_FG);
  float* FGMAP = (float*)alloc((size_t)NB * HW * 4);
  float* LSFB  = (float*)alloc((size_t)NB * HW * 4);
  int*   PEAKS = (int*)alloc((size_t)BK * 4);
  float* INDB  = (float*)alloc((size_t)BK * HW * 4);
  float* ATT1B = (float*)alloc((size_t)BK * HW * 4);
  float* SSUMB = (float*)alloc((size_t)NB * HW * 4);
  bf16*  WFG1  = (bf16*)alloc((size_t)CH * 25 * CH * 2);
  bf16*  WFG2  = (bf16*)alloc((size_t)CH * 25 * CH * 2);
  bf16*  WA1   = (bf16*)alloc((size_t)CP_A * 25 * CP_A * 2);
  bf16*  WA2   = (bf16*)alloc((size_t)CP_A * 25 * CP_A * 2);
  bf16*  REPP  = (bf16*)alloc(SZ_BIG);
  bf16*  ACTA  = (bf16*)alloc(SZ_BIG);
  bf16*  ACTB  = (bf16*)alloc(SZ_BIG);

  // Zero the padded frames (halo reads must see zeros). Graph-capturable.
  (void)hipMemsetAsync(XNP,  0, SZ_FG,  stream);
  (void)hipMemsetAsync(FGA1, 0, SZ_FG,  stream);
  (void)hipMemsetAsync(FGA2, 0, SZ_FG,  stream);
  (void)hipMemsetAsync(REPP, 0, SZ_BIG, stream);
  (void)hipMemsetAsync(ACTA, 0, SZ_BIG, stream);
  (void)hipMemsetAsync(ACTB, 0, SZ_BIG, stream);

  // ---- foreground LAN on x ------------------------------------------------
  pack_x_k<<<NB * HW * CH / 256, 256, 0, stream>>>(x, XNP);
  pack_w5_k<<<(CH * 25 * CH + 255) / 256, 256, 0, stream>>>(FG[2], WFG1, CH, CH, CH, CH);
  pack_w5_k<<<(CH * 25 * CH + 255) / 256, 256, 0, stream>>>(FG[4], WFG2, CH, CH, CH, CH);
  conv5x5_wmma<CH><<<dim3(NB * HW / 256, CH / 32), 256, 0, stream>>>(XNP,  WFG1, FG[3], FGA1, CH);
  conv5x5_wmma<CH><<<dim3(NB * HW / 256, CH / 32), 256, 0, stream>>>(FGA1, WFG2, FG[5], FGA2, CH);
  lan_tail<<<NB * HW / 256, 256, 0, stream>>>(FGA2, XNP, FG[6], FG[7], FG[0], FG[1],
                                              FG[8], FG[9], FGMAP, NB * HW, CH, CH);

  // ---- peaks / log-sigmoid / rep1 ----------------------------------------
  lsf_k<<<NB * HW / 256, 256, 0, stream>>>(FGMAP, LSFB, NB * HW);
  localmax_k<<<NB, HW, 0, stream>>>(FGMAP, PEAKS);
  ind_k<<<BK, HW, 0, stream>>>(PEAKS, INDB);
  rep1_k<<<BK * HW / 256, 256, 0, stream>>>(XNP, LSFB, INDB, REPP);

  // ---- attention LAN 1 ----------------------------------------------------
  pack_w5_k<<<(CP_A * 25 * CP_A + 255) / 256, 256, 0, stream>>>(A1[2], WA1, 258, 258, CP_A, CP_A);
  pack_w5_k<<<(CP_A * 25 * CP_A + 255) / 256, 256, 0, stream>>>(A1[4], WA2, 258, 258, CP_A, CP_A);
  conv5x5_wmma<CP_A><<<dim3(BK * HW / 256, CP_A / 32), 256, 0, stream>>>(REPP, WA1, A1[3], ACTA, 258);
  conv5x5_wmma<CP_A><<<dim3(BK * HW / 256, CP_A / 32), 256, 0, stream>>>(ACTA, WA2, A1[5], ACTB, 258);
  lan_tail<<<BK * HW / 256, 256, 0, stream>>>(ACTB, REPP, A1[6], A1[7], A1[0], A1[1],
                                              A1[8], A1[9], ATT1B, BK * HW, CP_A, 258);

  // ---- scope / rep2 -------------------------------------------------------
  scopesum_k<<<NB * HW / 256, 256, 0, stream>>>(ATT1B, objlen, SSUMB);
  rep2_k<<<BK * HW / 256, 256, 0, stream>>>(XNP, LSFB, ATT1B, SSUMB, REPP);

  // ---- attention LAN 2 ----------------------------------------------------
  pack_w5_k<<<(CP_A * 25 * CP_A + 255) / 256, 256, 0, stream>>>(A2[2], WA1, 259, 259, CP_A, CP_A);
  pack_w5_k<<<(CP_A * 25 * CP_A + 255) / 256, 256, 0, stream>>>(A2[4], WA2, 259, 259, CP_A, CP_A);
  conv5x5_wmma<CP_A><<<dim3(BK * HW / 256, CP_A / 32), 256, 0, stream>>>(REPP, WA1, A2[3], ACTA, 259);
  conv5x5_wmma<CP_A><<<dim3(BK * HW / 256, CP_A / 32), 256, 0, stream>>>(ACTA, WA2, A2[5], ACTB, 259);
  lan_tail<<<BK * HW / 256, 256, 0, stream>>>(ACTB, REPP, A2[6], A2[7], A2[0], A2[1],
                                              A2[8], A2[9], (float*)d_out, BK * HW, CP_A, 259);
}
